// MacroEncoder_89850715832643
// MI455X (gfx1250) — matrix operand order repaired and checked by reading, not compiled, not run on hardware
//
#include <hip/hip_runtime.h>
#include <math.h>

// ---------------- problem constants ----------------
constexpr int kND = 10000, kNP = 20000, kNS = 5000;
constexpr int kH = 8, kD = 16, kHID = 128, kIN = 256;
constexpr int kE_DD = 320000, kE_DP = 640000, kE_PD = 640000, kE_PP = 640000, kE_SD = 320000;
constexpr float kSLOPE = 0.2f;

// ---------------- WMMA types ----------------
typedef __bf16 bf16_t;
typedef bf16_t v16bf __attribute__((ext_vector_type(16)));
typedef float  v8f   __attribute__((ext_vector_type(8)));
union FragU { unsigned int u[8]; uint4 q[2]; v16bf v; };

__device__ __forceinline__ unsigned int f2bf(float f) {
  unsigned int u = __float_as_uint(f);
  u += 0x7FFFu + ((u >> 16) & 1u);          // round-to-nearest-even
  return u >> 16;
}
__device__ __forceinline__ unsigned int pack2bf(float lo, float hi) {
  return f2bf(lo) | (f2bf(hi) << 16);
}

// C[M,128] = A[M,K] @ B[K,128] (+bias). fp32 in/out, bf16 WMMA compute, fp32 acc.
// Block: 256 threads = 8 waves; tile 64 rows x 128 cols; K stepped by 32.
// LDS pitch 20 dwords: every row base is 16B-aligned (20*4B = 80B) so fragment
// reads stay ds_load_b128, and (20*lane mod 64)/4 = 5*lane mod 16 is a bank
// bijection across the 16 lanes of each half -> conflict-free.
constexpr int kLdsPitch = 20;

__global__ __launch_bounds__(256) void gemm128_bf16_wmma(
    const float* __restrict__ A, const float* __restrict__ B,
    const float* __restrict__ bias, float* __restrict__ C, int M, int K)
{
  __shared__ unsigned int As32[64 * kLdsPitch];    // [row][k/2] packed bf16 pairs
  __shared__ unsigned int Bs32[128 * kLdsPitch];   // [col][k/2] packed bf16 pairs
  const int tid  = threadIdx.x;
  const int lane = tid & 31;
  const int wave = tid >> 5;
  const int wm   = wave & 3;               // 16-row sub-tile
  const int wn   = wave >> 2;              // 64-col half
  const int row0 = blockIdx.x * 64;
  const int m    = lane & 15;
  const int hi   = lane >> 4;

  v8f acc[4];
  #pragma unroll
  for (int j = 0; j < 4; ++j)
    #pragma unroll
    for (int e = 0; e < 8; ++e) acc[j][e] = 0.0f;

  for (int k0 = 0; k0 < K; k0 += 32) {
    // A tile: 64 rows x 32 k = 512 float4; 2 per thread, b128 global loads,
    // branchless OOB (clamped row + zero fill), b64 LDS stores.
    #pragma unroll
    for (int it = 0; it < 2; ++it) {
      int idx = tid + it * 256;            // 0..511
      int r = idx >> 3, q = idx & 7;       // row in tile, float4 within row
      int gr = row0 + r;
      int grc = gr < M ? gr : (M - 1);
      float4 v = *(const float4*)(A + (long long)grc * K + k0 + q * 4);
      if (gr >= M) { v.x = 0.f; v.y = 0.f; v.z = 0.f; v.w = 0.f; }
      uint2 pk; pk.x = pack2bf(v.x, v.y); pk.y = pack2bf(v.z, v.w);
      *(uint2*)(As32 + r * kLdsPitch + q * 2) = pk;
    }
    // B tile: 32 k x 128 n, transposed to [n][k-pairs]; 512 tasks, 2 per thread.
    // Each task reads 4 cols from two adjacent k-rows (coalesced b128 pairs).
    #pragma unroll
    for (int it = 0; it < 2; ++it) {
      int idx = tid + it * 256;
      int kp = idx >> 5, n4 = idx & 31;    // k-pair 0..15, 4-col group 0..31
      const float* b0 = B + (long long)(k0 + 2 * kp) * kHID + n4 * 4;
      float4 x = *(const float4*)(b0);
      float4 y = *(const float4*)(b0 + kHID);
      int nb = n4 * 4;
      Bs32[(nb + 0) * kLdsPitch + kp] = pack2bf(x.x, y.x);
      Bs32[(nb + 1) * kLdsPitch + kp] = pack2bf(x.y, y.y);
      Bs32[(nb + 2) * kLdsPitch + kp] = pack2bf(x.z, y.z);
      Bs32[(nb + 3) * kLdsPitch + kp] = pack2bf(x.w, y.w);
    }
    __syncthreads();
    // A fragment (16x32 bf16, ISA 7.12.2): lane-half hi -> dwords {hi*4+0..3, hi*4+8..11}
    FragU fa;
    {
      const uint4* pa = (const uint4*)(As32 + (wm * 16 + m) * kLdsPitch + (hi << 2));
      fa.q[0] = pa[0];
      fa.q[1] = pa[2];
    }
    #pragma unroll
    for (int j = 0; j < 4; ++j) {
      int ncol = wn * 64 + j * 16 + m;
      // B fragment (32x16 bf16): lane-half hi -> dwords {hi*4+0..3, hi*4+8..11}
      FragU fb;
      const uint4* pb = (const uint4*)(Bs32 + ncol * kLdsPitch + (hi << 2));
      fb.q[0] = pb[0];
      fb.q[1] = pb[2];
      acc[j] = __builtin_amdgcn_wmma_f32_16x16x32_bf16(
          false, fa.v, false, fb.v, (short)0, acc[j], false, false);
    }
    __syncthreads();
  }
  // C/D layout: vgpr r -> row = r + 8*hi, col = lane%16
  #pragma unroll
  for (int j = 0; j < 4; ++j) {
    int col = wn * 64 + j * 16 + m;
    float bv = bias ? bias[col] : 0.0f;
    #pragma unroll
    for (int r = 0; r < 8; ++r) {
      int row = row0 + wm * 16 + hi * 8 + r;
      if (row < M) C[(long long)row * kHID + col] = acc[j][r] + bv;
    }
  }
}

// ---------------- elementwise / edge kernels ----------------
__global__ void fill_f32(float* __restrict__ p, float v, int n) {
  int t = blockIdx.x * blockDim.x + threadIdx.x;
  if (t < n) p[t] = v;
}

__global__ void copy_f32(const float* __restrict__ in, float* __restrict__ out, int n) {
  int t = blockIdx.x * blockDim.x + threadIdx.x;
  if (t < n) out[t] = in[t];
}

// out[n,h] = sum_d f[n, h*16+d] * a[h,d]
__global__ void head_dot(const float* __restrict__ f, const float* __restrict__ a,
                         float* __restrict__ out, int n) {
  int t = blockIdx.x * blockDim.x + threadIdx.x;
  if (t >= n * kH) return;
  int node = t >> 3, h = t & 7;
  const float* fp = f + (long long)node * kHID + h * kD;
  const float* ap = a + h * kD;
  float s = 0.0f;
  #pragma unroll
  for (int d = 0; d < kD; ++d) s += fp[d] * ap[d];
  out[t] = s;
}

// init stack[:, slot, :] = bias (fuses the GAT bias add; scatter accumulates on top)
__global__ void stack_bias_init(float* __restrict__ stack, const float* __restrict__ bias,
                                int N, int R, int slot) {
  int t = blockIdx.x * blockDim.x + threadIdx.x;
  if (t >= N * kHID) return;
  int n = t >> 7, c = t & 127;
  stack[((long long)n * R + slot) * kHID + c] = bias[c];
}

__device__ __forceinline__ void atomicMaxF(float* addr, float val) {
  if (val >= 0.0f) atomicMax((int*)addr, __float_as_int(val));
  else             atomicMin((unsigned int*)addr, __float_as_uint(val));
}

__device__ __forceinline__ float lrelu(float x) { return x > 0.0f ? x : kSLOPE * x; }

__global__ void edge_logit_max(const int* __restrict__ src, const int* __restrict__ dst,
                               const float* __restrict__ el, const float* __restrict__ er,
                               float* __restrict__ m, int E) {
  int t = blockIdx.x * blockDim.x + threadIdx.x;
  if (t >= E * kH) return;
  int e = t >> 3, h = t & 7;
  float x = lrelu(el[src[e] * kH + h] + er[dst[e] * kH + h]);
  atomicMaxF(&m[dst[e] * kH + h], x);
}

__global__ void edge_den(const int* __restrict__ src, const int* __restrict__ dst,
                         const float* __restrict__ el, const float* __restrict__ er,
                         const float* __restrict__ m, float* __restrict__ den, int E) {
  int t = blockIdx.x * blockDim.x + threadIdx.x;
  if (t >= E * kH) return;
  int e = t >> 3, h = t & 7;
  int d = dst[e];
  float x = lrelu(el[src[e] * kH + h] + er[d * kH + h]);
  float mm = m[d * kH + h]; if (mm < -1e29f) mm = 0.0f;   // isfinite guard
  atomicAdd(&den[d * kH + h], expf(x - mm));
}

// stack[dst, slot, c0..c0+3] += alpha(e,h) * fs[src, c0..c0+3]
// 4 channels per thread: amortizes src/dst/logit loads 4x, fs gather becomes b128.
__global__ void edge_scatter(const int* __restrict__ src, const int* __restrict__ dst,
                             const float* __restrict__ el, const float* __restrict__ er,
                             const float* __restrict__ m, const float* __restrict__ den,
                             const float* __restrict__ fs, float* __restrict__ stack,
                             int E, int R, int slot) {
  long long t = (long long)blockIdx.x * blockDim.x + threadIdx.x;
  if (t >= (long long)E * 32) return;
  int e = (int)(t >> 5), g = (int)(t & 31);
  int c0 = g * 4, h = c0 >> 4;
  int s = src[e], d = dst[e];
  float x = lrelu(el[s * kH + h] + er[d * kH + h]);
  float mm = m[d * kH + h]; if (mm < -1e29f) mm = 0.0f;
  float dn = den[d * kH + h]; if (dn < 1e-9f) dn = 1e-9f;
  float alpha = expf(x - mm) / dn;
  float4 v = *(const float4*)(fs + (long long)s * kHID + c0);
  float* o = stack + ((long long)d * R + slot) * kHID + c0;
  atomicAdd(o + 0, alpha * v.x);
  atomicAdd(o + 1, alpha * v.y);
  atomicAdd(o + 2, alpha * v.z);
  atomicAdd(o + 3, alpha * v.w);
}

// semantic attention: s_r = tanh(t[n,r,:]) . W2 + b2 ; softmax over r ; out = sum w_r f_r
__global__ __launch_bounds__(128) void semantic_attn(
    const float* __restrict__ stack, const float* __restrict__ tproj,
    const float* __restrict__ W2, const float* __restrict__ b2,
    float* __restrict__ out, int N, int R) {
  __shared__ float red[128];
  __shared__ float sc[4];
  int n = blockIdx.x, c = threadIdx.x;
  for (int r = 0; r < R; ++r) {
    red[c] = tanhf(tproj[((long long)n * R + r) * kHID + c]) * W2[c];
    __syncthreads();
    for (int s = 64; s > 0; s >>= 1) {
      if (c < s) red[c] += red[c + s];
      __syncthreads();
    }
    if (c == 0) sc[r] = red[0] + b2[0];
    __syncthreads();
  }
  float mx = sc[0];
  for (int r = 1; r < R; ++r) mx = fmaxf(mx, sc[r]);
  float wsum = 0.0f, wv[3];
  for (int r = 0; r < R; ++r) { wv[r] = expf(sc[r] - mx); wsum += wv[r]; }
  float o = 0.0f;
  for (int r = 0; r < R; ++r)
    o += (wv[r] / wsum) * stack[((long long)n * R + r) * kHID + c];
  out[(long long)n * kHID + c] = o;
}

// ---------------- host orchestration ----------------
struct RelP { const float *W, *al, *ar, *bias; };
struct SemP { const float *W1, *W2, *b1, *b2; };

static inline unsigned cdivu(long long a, long long b) { return (unsigned)((a + b - 1) / b); }

extern "C" void kernel_launch(void* const* d_in, const int* in_sizes, int n_in,
                              void* d_out, int out_size, void* d_ws, size_t ws_size,
                              hipStream_t stream) {
  (void)in_sizes; (void)n_in; (void)out_size; (void)ws_size;
  const float* feat_d = (const float*)d_in[0];
  const float* feat_p = (const float*)d_in[1];
  const float* feat_s = (const float*)d_in[2];
  const int* edd_s = (const int*)d_in[3];  const int* edd_d = (const int*)d_in[4];
  const int* edp_s = (const int*)d_in[5];  const int* edp_d = (const int*)d_in[6];
  const int* epd_s = (const int*)d_in[7];  const int* epd_d = (const int*)d_in[8];
  const int* epp_s = (const int*)d_in[9];  const int* epp_d = (const int*)d_in[10];
  const int* esd_s = (const int*)d_in[11]; const int* esd_d = (const int*)d_in[12];

  // params leaves in JAX tree-flatten order (sorted dict keys): "layers" < "proj".
  // layer: rel{drug2drug,drug2protein,protein2drug,protein2protein,sideeffect2drug}
  //        each {W,al,ar,bias}; then sem {W1,W2,b1,b2}. proj: drug{W,b},protein{W,b},sideeffect{W,b}.
  auto F = [&](int i) { return (const float*)d_in[i]; };
  RelP rel[2][5]; SemP sem[2];
  for (int L = 0; L < 2; ++L) {
    int lb = 13 + L * 24;
    for (int r = 0; r < 5; ++r)
      rel[L][r] = { F(lb + 4*r), F(lb + 4*r + 1), F(lb + 4*r + 2), F(lb + 4*r + 3) };
    sem[L] = { F(lb + 20), F(lb + 21), F(lb + 22), F(lb + 23) };
  }
  int pb = 13 + 48;
  const float *projWd = F(pb+0), *projbd = F(pb+1);
  const float *projWp = F(pb+2), *projbp = F(pb+3);
  const float *projWs = F(pb+4), *projbs = F(pb+5);

  // workspace layout (floats); total ~28.2M floats ~= 113 MB
  float* ws = (float*)d_ws;
  long long off = 0;
  float* h_d  = ws + off; off += (long long)kND * kHID;
  float* h_p  = ws + off; off += (long long)kNP * kHID;
  float* h_s  = ws + off; off += (long long)kNS * kHID;
  float* h_d2 = ws + off; off += (long long)kND * kHID;
  float* h_p2 = ws + off; off += (long long)kNP * kHID;
  float* dstk = ws + off; off += (long long)kND * 3 * kHID;
  float* pstk = ws + off; off += (long long)kNP * 2 * kHID;
  float* fsb  = ws + off; off += (long long)kNP * kHID;
  float* fdb  = ws + off; off += (long long)kNP * kHID;
  float* elb  = ws + off; off += (long long)kNP * kH;
  float* erb  = ws + off; off += (long long)kNP * kH;
  float* mb   = ws + off; off += (long long)kNP * kH;
  float* denb = ws + off; off += (long long)kNP * kH;
  float* tbuf = ws + off; off += (long long)kNP * 2 * kHID;

  auto gemm = [&](const float* A, const float* B, const float* bias, float* C, int M, int K) {
    gemm128_bf16_wmma<<<dim3(cdivu(M, 64)), dim3(256), 0, stream>>>(A, B, bias, C, M, K);
  };

  // node-type projections (K = 256)
  gemm(feat_d, projWd, projbd, h_d, kND, kIN);
  gemm(feat_p, projWp, projbp, h_p, kNP, kIN);
  gemm(feat_s, projWs, projbs, h_s, kNS, kIN);

  auto run_rel = [&](const float* hsrc, int Ns, const float* hdst, int Nd,
                     const int* src, const int* dst, int E,
                     const RelP& rp, float* stack, int R, int slot, bool same) {
    gemm(hsrc, rp.W, nullptr, fsb, Ns, kHID);
    const float* fd = fsb;
    if (!same) { gemm(hdst, rp.W, nullptr, fdb, Nd, kHID); fd = fdb; }
    head_dot<<<cdivu((long long)Ns * kH, 256), 256, 0, stream>>>(fsb, rp.al, elb, Ns);
    head_dot<<<cdivu((long long)Nd * kH, 256), 256, 0, stream>>>(fd,  rp.ar, erb, Nd);
    fill_f32<<<cdivu((long long)Nd * kH, 256), 256, 0, stream>>>(mb, -1e30f, Nd * kH);
    fill_f32<<<cdivu((long long)Nd * kH, 256), 256, 0, stream>>>(denb, 0.0f, Nd * kH);
    stack_bias_init<<<cdivu((long long)Nd * kHID, 256), 256, 0, stream>>>(stack, rp.bias, Nd, R, slot);
    edge_logit_max<<<cdivu((long long)E * kH, 256), 256, 0, stream>>>(src, dst, elb, erb, mb, E);
    edge_den<<<cdivu((long long)E * kH, 256), 256, 0, stream>>>(src, dst, elb, erb, mb, denb, E);
    edge_scatter<<<cdivu((long long)E * 32, 256), 256, 0, stream>>>(
        src, dst, elb, erb, mb, denb, fsb, stack, E, R, slot);
  };

  auto run_sem = [&](const float* stack, int N, int R, const SemP& sp, float* out) {
    gemm(stack, sp.W1, sp.b1, tbuf, N * R, kHID);   // stack is contiguous [N*R, 128]
    semantic_attn<<<dim3((unsigned)N), dim3(128), 0, stream>>>(stack, tbuf, sp.W2, sp.b2, out, N, R);
  };

  const float* hd = h_d;
  const float* hp = h_p;
  float* outD = (float*)d_out;
  float* outP = (float*)d_out + (long long)kND * kHID;
  float* outS = (float*)d_out + (long long)(kND + kNP) * kHID;

  for (int L = 0; L < 2; ++L) {
    float* od = (L == 0) ? h_d2 : outD;
    float* op = (L == 0) ? h_p2 : outP;
    // drug destinations: stack slots [drug2drug, protein2drug, sideeffect2drug]
    run_rel(hd,  kND, hd, kND, edd_s, edd_d, kE_DD, rel[L][0], dstk, 3, 0, true);
    run_rel(hp,  kNP, hd, kND, epd_s, epd_d, kE_PD, rel[L][2], dstk, 3, 1, false);
    run_rel(h_s, kNS, hd, kND, esd_s, esd_d, kE_SD, rel[L][4], dstk, 3, 2, false);
    // protein destinations: stack slots [drug2protein, protein2protein]
    run_rel(hd,  kND, hp, kNP, edp_s, edp_d, kE_DP, rel[L][1], pstk, 2, 0, false);
    run_rel(hp,  kNP, hp, kNP, epp_s, epp_d, kE_PP, rel[L][3], pstk, 2, 1, true);
    run_sem(dstk, kND, 3, sem[L], od);
    run_sem(pstk, kNP, 2, sem[L], op);
    hd = od; hp = op;
  }
  // sideeffect nodes never receive a relation: raw projection output
  copy_f32<<<cdivu((long long)kNS * kHID, 256), 256, 0, stream>>>(h_s, outS, kNS * kHID);
}
